// MultiHeadAttention_24807731101915
// MI455X (gfx1250) — matrix operand order repaired and checked by reading, not compiled, not run on hardware
//
#include <hip/hip_runtime.h>
#include <math.h>

// ---------------------------------------------------------------------------
// MHA forward for MI455X (gfx1250, wave32): bf16 WMMA + TDM async LDS staging.
//   x:[4,2048,1024] f32; wq/wk/wv/wo:[1024,1024] f32; biases [1024] f32.
// Pipeline:
//   1) cvt f32->bf16 (x + weights) into workspace
//   2) WMMA GEMM (TDM-fed, double-buffered LDS):
//        Q = x@wq -> [B,H,S,64] bf16 ; K = x@wk -> [B,H,S,64] bf16
//        V = x@wv -> [B,H,64,S] bf16 (transposed so PV B-fragments are linear)
//   3) flash-attention (streaming softmax, prefetched K/V) -> [B,S,1024] bf16
//   4) WMMA GEMM: out = attn@wo + bo -> f32
// ---------------------------------------------------------------------------

typedef __bf16 bf16_t;
typedef __attribute__((ext_vector_type(16))) __bf16 v16bf;
typedef __attribute__((ext_vector_type(8)))  __bf16 v8bf;
typedef __attribute__((ext_vector_type(4)))  __bf16 v4bf;
typedef __attribute__((ext_vector_type(8)))  float  v8f;
typedef __attribute__((ext_vector_type(4)))  unsigned int u32x4;
typedef __attribute__((ext_vector_type(8)))  unsigned int u32x8;

#define D_MODEL 1024
#define S_LEN   2048
#define BATCH   4
#define HEADS   16
#define DEPTH   64
#define M_TOT   (BATCH * S_LEN)   /* 8192 rows in the big GEMMs */

__device__ __forceinline__ v16bf ld_frag(const bf16_t* p0, const bf16_t* p1) {
  union { v16bf v; v8bf h[2]; } u;
  u.h[0] = *reinterpret_cast<const v8bf*>(p0);
  u.h[1] = *reinterpret_cast<const v8bf*>(p1);
  return u.v;
}

__device__ __forceinline__ v8f wmma_bf16(v16bf a, v16bf b, v8f c) {
  return __builtin_amdgcn_wmma_f32_16x16x32_bf16(
      false, a, false, b, (short)0, c, false, false);
}

// ---------------------------------------------------------------------------
// TDM: DMA a 16-row x 32-col bf16 sub-tile (row stride D_MODEL elements in
// global) into LDS with 4-DWORD padding after every 16 DWORDs (=> 40-element
// padded LDS rows). 2D descriptor, groups 0+1 only (ISA ch.8).
// ---------------------------------------------------------------------------
__device__ __forceinline__ void tdm_load_a16x32(const bf16_t* gsrc,
                                                unsigned lds_addr) {
  unsigned long long ga = (unsigned long long)(size_t)gsrc;
  u32x4 g0;
  g0[0] = 1u;                                            // count=1, user mode
  g0[1] = lds_addr;                                      // LDS byte address
  g0[2] = (unsigned)ga;                                  // global_addr[31:0]
  g0[3] = (unsigned)((ga >> 32) & 0x1FFFFFFull) | (2u << 30); // addr hi | type=2
  u32x8 g1;
  g1[0] = (1u << 16)      // data_size = 1 -> 2 bytes
        | (1u << 20)      // pad_enable
        | (3u << 22)      // pad_interval code 3 -> every 16 DWORDs (1 row)
        | (3u << 25);     // pad_amount  code 3 -> 4 DWORDs (16B -> 40-elem rows)
  g1[1] = (unsigned)(D_MODEL & 0xFFFF) << 16;            // tensor_dim0 lo16
  g1[2] = (unsigned)(D_MODEL >> 16)
        | ((unsigned)(M_TOT & 0xFFFF) << 16);            // dim0 hi | dim1 lo
  g1[3] = (unsigned)(M_TOT >> 16) | (32u << 16);         // dim1 hi | tile_dim0=32
  g1[4] = 16u;                                           // tile_dim1=16, dim2=0
  g1[5] = (unsigned)D_MODEL;                             // tensor_dim0_stride
  g1[6] = 0u;
  g1[7] = 0u;
  asm volatile("tensor_load_to_lds %0, %1" :: "s"(g0), "s"(g1) : "memory");
}

// ---------------------------------------------------------------------------
// Kernel 1: f32 -> bf16 conversion (vectorized x4)
// ---------------------------------------------------------------------------
__global__ void cvt_f32_to_bf16(const float* __restrict__ src,
                                bf16_t* __restrict__ dst, int n4) {
  int i = blockIdx.x * blockDim.x + threadIdx.x;
  if (i >= n4) return;
  float4 f = reinterpret_cast<const float4*>(src)[i];
  v4bf o;
  o.x = (__bf16)f.x; o.y = (__bf16)f.y; o.z = (__bf16)f.z; o.w = (__bf16)f.w;
  reinterpret_cast<v4bf*>(dst)[i] = o;
}

// ---------------------------------------------------------------------------
// Stage one K-step worth of tiles into LDS buffer `buf`:
//  - A tile 128x32: each wave issues one TDM op for its 16-row slice
//  - W tile 32x256: cooperative load + transpose into Bs[n][k]
// ---------------------------------------------------------------------------
__device__ __forceinline__ void stage_tiles(const bf16_t* __restrict__ A,
                                            const bf16_t* __restrict__ W,
                                            int Mbase, int Nbase, int k0,
                                            int tid, int wave,
                                            bf16_t (*As)[40],
                                            bf16_t (*Bs)[40]) {
  // --- A via TDM (async, tracked by TENSORcnt) ---------------------------
  tdm_load_a16x32(&A[(size_t)(Mbase + wave * 16) * D_MODEL + k0],
                  (unsigned)(size_t)&As[wave * 16][0]);
  // --- W transposed via VGPRs -------------------------------------------
#pragma unroll
  for (int i = 0; i < 4; ++i) {
    int c  = tid + 256 * i;              // 0..1023 chunks of 8 along n
    int kk = c >> 5;                     // 0..31
    int n  = (c & 31) * 8;               // 0..248
    v8bf wv8 = *reinterpret_cast<const v8bf*>(
        &W[(size_t)(k0 + kk) * D_MODEL + Nbase + n]);
#pragma unroll
    for (int j = 0; j < 8; ++j) Bs[n + j][kk] = wv8[j];
  }
}

// ---------------------------------------------------------------------------
// Kernel 2: bf16 WMMA GEMM  C[M=8192, N=1024] = A @ W  (K = 1024)
// 256 thr = 8 waves; block tile 128x256; wave tile 64x64; double-buffered LDS.
// mode 0/1: bf16 Q/K [B,H,S,64]; mode 2: bf16 V^T [B,H,64,S];
// mode 3: f32 [M,1024] + bias.
// ---------------------------------------------------------------------------
__global__ __launch_bounds__(256)
void gemm_bf16_kernel(const bf16_t* __restrict__ A,
                      const bf16_t* __restrict__ W,
                      const float* __restrict__ bias,
                      void* __restrict__ out, int mode) {
  __shared__ __align__(16) bf16_t As[2][128][40];
  __shared__ __align__(16) bf16_t Bs[2][256][40];

  const int tid   = threadIdx.x;
  const int Mbase = blockIdx.y * 128;
  const int Nbase = blockIdx.x * 256;
  const int wave  = tid >> 5, lane = tid & 31;
  const int wm    = wave & 1, wn = wave >> 1;    // 2 x 4 wave grid
  const int half  = lane >> 4, lc = lane & 15;

  v8f acc[4][4];
#pragma unroll
  for (int i = 0; i < 4; ++i)
#pragma unroll
    for (int j = 0; j < 4; ++j)
      acc[i][j] = (v8f){0.f, 0.f, 0.f, 0.f, 0.f, 0.f, 0.f, 0.f};

  // prologue: stage tile 0 into buffer 0
  stage_tiles(A, W, Mbase, Nbase, 0, tid, wave, As[0], Bs[0]);

  int buf = 0;
  for (int k0 = 0; k0 < D_MODEL; k0 += 32) {
    __builtin_amdgcn_s_wait_tensorcnt(0);   // TDM for `buf` complete (per wave)
    __syncthreads();                        // B stores visible, waves aligned
    if (k0 + 32 < D_MODEL)                  // overlap next tile DMA with WMMA
      stage_tiles(A, W, Mbase, Nbase, k0 + 32, tid, wave, As[buf ^ 1], Bs[buf ^ 1]);

    v16bf af[4], bfr[4];
#pragma unroll
    for (int sm = 0; sm < 4; ++sm) {
      const bf16_t* p = &As[buf][wm * 64 + sm * 16 + lc][half * 8];
      af[sm] = ld_frag(p, p + 16);          // K = h*8+0..7 and 16+h*8+0..7
    }
#pragma unroll
    for (int sn = 0; sn < 4; ++sn) {
      const bf16_t* p = &Bs[buf][wn * 64 + sn * 16 + lc][half * 8];
      bfr[sn] = ld_frag(p, p + 16);         // K = h*8+0..7 and 16+h*8+0..7
    }
#pragma unroll
    for (int sm = 0; sm < 4; ++sm)
#pragma unroll
      for (int sn = 0; sn < 4; ++sn)
        acc[sm][sn] = wmma_bf16(af[sm], bfr[sn], acc[sm][sn]);

    buf ^= 1;
  }

  // --- epilogue: C element e of lane = (row e+8*half, col lc) -------------
#pragma unroll
  for (int sm = 0; sm < 4; ++sm)
#pragma unroll
    for (int sn = 0; sn < 4; ++sn)
#pragma unroll
      for (int e = 0; e < 8; ++e) {
        int gm = Mbase + wm * 64 + sm * 16 + e + 8 * half;
        int gn = Nbase + wn * 64 + sn * 16 + lc;
        float v = acc[sm][sn][e] + bias[gn];
        if (mode == 3) {
          reinterpret_cast<float*>(out)[(size_t)gm * D_MODEL + gn] = v;
        } else {
          int b = gm >> 11, s = gm & (S_LEN - 1);
          int h = gn >> 6,  d = gn & (DEPTH - 1);
          bf16_t* o = reinterpret_cast<bf16_t*>(out);
          if (mode == 2)   // V^T: [B,H,64,S]
            o[((size_t)(b * HEADS + h) * DEPTH + d) * S_LEN + s] = (bf16_t)v;
          else             // Q/K: [B,H,S,64]
            o[((size_t)(b * HEADS + h) * S_LEN + s) * DEPTH + d] = (bf16_t)v;
        }
      }
}

// ---------------------------------------------------------------------------
// Kernel 3: flash attention.
// Grid (S/64, B*H); block 128 = 4 waves. Wave w owns q rows [qt*64+w*16, +16).
// Streams keys 32 at a time: 4 WMMA (QK^T) + online softmax + 4 WMMA (PV),
// prefetching the next K/V tiles (L2-resident) each iteration.
// ---------------------------------------------------------------------------
__global__ __launch_bounds__(128)
void flash_attn_kernel(const bf16_t* __restrict__ Q,
                       const bf16_t* __restrict__ K,
                       const bf16_t* __restrict__ VT,
                       bf16_t* __restrict__ AO) {
  __shared__ __align__(16) bf16_t Pst[4][16][40];   // per-wave P tile (16x32)

  const int qt   = blockIdx.x;
  const int bh   = blockIdx.y;
  const int wave = threadIdx.x >> 5, lane = threadIdx.x & 31;
  const int half = lane >> 4, lc = lane & 15;
  const float scale = 0.125f;               // 1/sqrt(64)

  const bf16_t* qptr =
      Q + ((size_t)bh * S_LEN + qt * 64 + wave * 16 + lc) * DEPTH;
  v16bf qf[2];
  qf[0] = ld_frag(qptr + half * 8,        qptr + 16 + half * 8);   // d 0..31
  qf[1] = ld_frag(qptr + 32 + half * 8,   qptr + 48 + half * 8);   // d 32..63

  v8f o[4];
#pragma unroll
  for (int i = 0; i < 4; ++i)
    o[i] = (v8f){0.f, 0.f, 0.f, 0.f, 0.f, 0.f, 0.f, 0.f};
  float mi[8], li[8];
#pragma unroll
  for (int e = 0; e < 8; ++e) { mi[e] = -INFINITY; li[e] = 0.f; }

  for (int s0 = 0; s0 < S_LEN; s0 += 32) {
    // pull the next K/V tiles toward the WGP while we compute this one
    if (s0 + 32 < S_LEN) {
      __builtin_prefetch(K + ((size_t)bh * S_LEN + s0 + 32 + lc) * DEPTH, 0, 1);
      __builtin_prefetch(VT + ((size_t)bh * DEPTH + lc) * S_LEN + s0 + 32, 0, 1);
    }
    // ---- logits 16x32: two 16x16 C tiles ---------------------------------
    v8f lg[2];
#pragma unroll
    for (int kt = 0; kt < 2; ++kt) {
      lg[kt] = (v8f){0.f, 0.f, 0.f, 0.f, 0.f, 0.f, 0.f, 0.f};
      const bf16_t* kb =
          K + ((size_t)bh * S_LEN + s0 + kt * 16 + lc) * DEPTH + half * 8;
      v16bf kf0 = ld_frag(kb,      kb + 16);       // depth 0..31
      v16bf kf1 = ld_frag(kb + 32, kb + 48);       // depth 32..63
      lg[kt] = wmma_bf16(qf[0], kf0, lg[kt]);
      lg[kt] = wmma_bf16(qf[1], kf1, lg[kt]);
    }

    // ---- online softmax (row = e + 8*half; 16 lanes of a half share row) -
    float mnew[8], corr[8];
#pragma unroll
    for (int e = 0; e < 8; ++e) {
      float v = fmaxf(lg[0][e], lg[1][e]) * scale;
#pragma unroll
      for (int m = 1; m < 16; m <<= 1) v = fmaxf(v, __shfl_xor(v, m, 32));
      mnew[e] = fmaxf(mi[e], v);
      corr[e] = __expf(mi[e] - mnew[e]);
    }
#pragma unroll
    for (int e = 0; e < 8; ++e) {
      float p0 = __expf(lg[0][e] * scale - mnew[e]);
      float p1 = __expf(lg[1][e] * scale - mnew[e]);
      lg[0][e] = p0; lg[1][e] = p1;
      float rs = p0 + p1;
#pragma unroll
      for (int m = 1; m < 16; m <<= 1) rs += __shfl_xor(rs, m, 32);
      li[e] = li[e] * corr[e] + rs;
      mi[e] = mnew[e];
    }
#pragma unroll
    for (int dn = 0; dn < 4; ++dn)
#pragma unroll
      for (int e = 0; e < 8; ++e) o[dn][e] *= corr[e];

    // ---- restripe P (C layout) -> A-fragment layout via LDS --------------
#pragma unroll
    for (int kt = 0; kt < 2; ++kt)
#pragma unroll
      for (int e = 0; e < 8; ++e)
        Pst[wave][e + 8 * half][kt * 16 + lc] = (bf16_t)lg[kt][e];
    __syncthreads();
    const bf16_t* pl = &Pst[wave][lc][half * 8];
    v16bf pf = ld_frag(pl, pl + 16);

    // ---- O += P(16x32) x V(32x64): 4 WMMA over depth sub-tiles -----------
#pragma unroll
    for (int dn = 0; dn < 4; ++dn) {
      const bf16_t* vb =
          VT + ((size_t)bh * DEPTH + dn * 16 + lc) * S_LEN + s0 + half * 8;
      v16bf vf = ld_frag(vb, vb + 16);
      o[dn] = wmma_bf16(pf, vf, o[dn]);
    }
    __syncthreads();
  }

  // ---- epilogue: normalize and write [B,S,1024] bf16 ---------------------
  float inv[8];
#pragma unroll
  for (int e = 0; e < 8; ++e) inv[e] = 1.0f / li[e];
  const int b = bh >> 4, h = bh & (HEADS - 1);
#pragma unroll
  for (int dn = 0; dn < 4; ++dn)
#pragma unroll
    for (int e = 0; e < 8; ++e) {
      int row = qt * 64 + wave * 16 + e + 8 * half;
      AO[((size_t)(b * S_LEN + row)) * D_MODEL + h * DEPTH + dn * 16 + lc] =
          (bf16_t)(o[dn][e] * inv[e]);
    }
}

// ---------------------------------------------------------------------------
// Host: orchestrate the pipeline on `stream`.
// ---------------------------------------------------------------------------
extern "C" void kernel_launch(void* const* d_in, const int* in_sizes, int n_in,
                              void* d_out, int out_size, void* d_ws, size_t ws_size,
                              hipStream_t stream) {
  const float* x  = (const float*)d_in[0];
  const float* wq = (const float*)d_in[1];
  const float* bq = (const float*)d_in[2];
  const float* wk = (const float*)d_in[3];
  const float* bk = (const float*)d_in[4];
  const float* wv = (const float*)d_in[5];
  const float* bv = (const float*)d_in[6];
  const float* wo = (const float*)d_in[7];
  const float* bo = (const float*)d_in[8];

  char* ws = (char*)d_ws;
  size_t off = 0;
  const size_t XB_BYTES = (size_t)M_TOT * D_MODEL * sizeof(bf16_t);   // 16 MB
  const size_t W_BYTES  = (size_t)D_MODEL * D_MODEL * sizeof(bf16_t); //  2 MB
  const size_t T_BYTES  = (size_t)BATCH * HEADS * S_LEN * DEPTH * sizeof(bf16_t);

  bf16_t* xb   = (bf16_t*)(ws + off); off += XB_BYTES;
  bf16_t* wqb  = (bf16_t*)(ws + off); off += W_BYTES;
  bf16_t* wkb  = (bf16_t*)(ws + off); off += W_BYTES;
  bf16_t* wvb  = (bf16_t*)(ws + off); off += W_BYTES;
  bf16_t* wob  = (bf16_t*)(ws + off); off += W_BYTES;
  bf16_t* Qb   = (bf16_t*)(ws + off); off += T_BYTES;
  bf16_t* Kb   = (bf16_t*)(ws + off); off += T_BYTES;
  bf16_t* VTb  = (bf16_t*)(ws + off); off += T_BYTES;
  bf16_t* AOb  = (bf16_t*)(ws + off); off += XB_BYTES;

  {
    int n4 = (M_TOT * D_MODEL) / 4;
    cvt_f32_to_bf16<<<(n4 + 255) / 256, 256, 0, stream>>>(x, xb, n4);
    int w4 = (D_MODEL * D_MODEL) / 4;
    cvt_f32_to_bf16<<<(w4 + 255) / 256, 256, 0, stream>>>(wq, wqb, w4);
    cvt_f32_to_bf16<<<(w4 + 255) / 256, 256, 0, stream>>>(wk, wkb, w4);
    cvt_f32_to_bf16<<<(w4 + 255) / 256, 256, 0, stream>>>(wv, wvb, w4);
    cvt_f32_to_bf16<<<(w4 + 255) / 256, 256, 0, stream>>>(wo, wob, w4);
  }

  dim3 ggrid(D_MODEL / 256, M_TOT / 128);   // (4, 64)
  gemm_bf16_kernel<<<ggrid, 256, 0, stream>>>(xb, wqb, bq, Qb, 0);
  gemm_bf16_kernel<<<ggrid, 256, 0, stream>>>(xb, wkb, bk, Kb, 1);
  gemm_bf16_kernel<<<ggrid, 256, 0, stream>>>(xb, wvb, bv, VTb, 2);

  dim3 agrid(S_LEN / 64, BATCH * HEADS);    // (32, 64)
  flash_attn_kernel<<<agrid, 128, 0, stream>>>(Qb, Kb, VTb, AOb);

  gemm_bf16_kernel<<<ggrid, 256, 0, stream>>>(AOb, wob, bo, d_out, 3);
}
